// HPNLearner_65120294142226
// MI455X (gfx1250) — compile-verified, matrix operand order checked by reading, not run on hardware
//
#include <hip/hip_runtime.h>

// ---------------------------------------------------------------------------
// Types for CDNA5 WMMA (wave32).
// ---------------------------------------------------------------------------
typedef _Float16 half_t;
typedef half_t v16h __attribute__((ext_vector_type(16)));
typedef float  v8f  __attribute__((ext_vector_type(8)));

union H2U { half_t h[2]; unsigned u; };

__device__ inline unsigned pack2h(float a, float b) {
  H2U t; t.h[0] = (half_t)a; t.h[1] = (half_t)b; return t.u;
}

// All tensors in this network are < 2^31 elements -> 32-bit indexing everywhere.
#define GS_LOOP(i, total)                                                      \
  for (unsigned i = blockIdx.x * blockDim.x + threadIdx.x; i < (total);        \
       i += gridDim.x * blockDim.x)

// ---------------------------------------------------------------------------
// Generic 7-D gather / scatter-add:  out[flat] (+)= in[dot(idx, strides)]
// ---------------------------------------------------------------------------
__global__ void gather7_kernel(const float* __restrict__ in, float* __restrict__ out,
                               unsigned total, int d1, int d2, int d3, int d4,
                               int d5, int d6, int s0, int s1, int s2, int s3,
                               int s4, int s5, int s6, int accumulate) {
  GS_LOOP(idx, total) {
    unsigned t = idx;
    int i6 = (int)(t % (unsigned)d6); t /= (unsigned)d6;
    int i5 = (int)(t % (unsigned)d5); t /= (unsigned)d5;
    int i4 = (int)(t % (unsigned)d4); t /= (unsigned)d4;
    int i3 = (int)(t % (unsigned)d3); t /= (unsigned)d3;
    int i2 = (int)(t % (unsigned)d2); t /= (unsigned)d2;
    int i1 = (int)(t % (unsigned)d1); t /= (unsigned)d1;
    int i0 = (int)t;
    int off = i0 * s0 + i1 * s1 + i2 * s2 + i3 * s3 + i4 * s4 + i5 * s5 + i6 * s6;
    float v = in[off];
    if (accumulate) out[idx] += v; else out[idx] = v;
  }
}

// ---------------------------------------------------------------------------
// kmeta build: per conv launch, precompute k -> (x-offset, kh|kw) so the conv
// hot loop carries no integer divisions.
//   km[k] = { cin*HW + kh*W + kw,  (kh<<16)|kw }
// ---------------------------------------------------------------------------
__global__ void kmeta_build_kernel(int2* __restrict__ km, int Kdim, int KK,
                                   int K, int HW, int W) {
  GS_LOOP(k, (unsigned)Kdim) {
    int cin = (int)k / KK;
    int r = (int)k - cin * KK;
    int kh = r / K, kw = r - (r / K) * K;
    int2 v;
    v.x = cin * HW + kh * W + kw;
    v.y = (kh << 16) | kw;
    km[k] = v;
  }
}

// ---------------------------------------------------------------------------
// Implicit-GEMM conv2d on WMMA.  y(B,Cout,OH,OW) = conv(x(B,Cin,H,W), w, bias)
// GEMM: M=Cout, N=B*OH*OW, K=Cin*KH*KW.  One wave32 per 16x16 C tile.
// 64-wide K slab -> two v_wmma_f32_16x16x32_f16 per loop body.
// A staged via float4 fast path; B staged column-owner into transposed LDS so
// both fragment operands load as ds_load_b128.  k-decompose read from the
// precomputed kmeta table (global_load_b64).  All loads branchless/clamped.
// ---------------------------------------------------------------------------
__global__ __launch_bounds__(32) void conv2d_wmma_kernel(
    const float* __restrict__ x, const float* __restrict__ w,
    const float* __restrict__ bias, float* __restrict__ y,
    const int2* __restrict__ km, int B, int Cin, int H, int W, int Cout, int K,
    int stride, int pad, int OH, int OW, int accumulate, int do_relu) {
  __shared__ half_t As[16][64];      // [m][k]
  __shared__ half_t Bt[16][64 + 8];  // [n][k] transposed; 144B pitch (16B mult)
  __shared__ int kmeta_off[64];      // cin*HW + kh*W + kw
  __shared__ int kmeta_hw[64];       // (kh<<16)|kw ; -1 marks k >= Kdim
  const int lane = (int)threadIdx.x;
  const int tile_n = (int)blockIdx.x, tile_m = (int)blockIdx.y;
  const int KK = K * K;
  const int Kdim = Cin * KK;
  const int Ndim = B * OH * OW;
  const int OHW = OH * OW;

  // ---- this lane's output-column metadata (column-owner scheme)
  const int mycol = lane & 15;
  const int khalf = lane >> 4;  // which 32-k half of the slab this lane stages
  int col_off, col_ih0, col_iw0, col_v;
  {
    int n = tile_n * 16 + mycol;
    col_v = (n < Ndim) ? 1 : 0;
    int nc = col_v ? n : 0;
    int b = nc / OHW;
    int r2 = nc - b * OHW;
    int oh = r2 / OW, ow = r2 - (r2 / OW) * OW;
    col_ih0 = oh * stride - pad;
    col_iw0 = ow * stride - pad;
    col_off = b * Cin * H * W + col_ih0 * W + col_iw0;
  }

  v8f acc = {0.f, 0.f, 0.f, 0.f, 0.f, 0.f, 0.f, 0.f};
  const int arow = tile_m * 16 + (lane >> 1);
  const int mvld = (arow < Cout) ? 1 : 0;
  const float* wrow = w + (mvld ? arow : 0) * Kdim;
  const int k4ok = ((Kdim & 3) == 0) ? 1 : 0;  // float4 alignment guard

  for (int k0 = 0; k0 < Kdim; k0 += 64) {
    __syncthreads();  // protect LDS overwrite (WAR vs previous fragments)
    // ---- slab kmeta: 2 entries per lane from the precomputed table
#pragma unroll
    for (int u = 0; u < 2; ++u) {
      int kk = lane + u * 32;
      int k = k0 + kk;
      int kvalid = (k < Kdim) ? 1 : 0;
      int2 mv = km[kvalid ? k : 0];
      kmeta_off[kk] = mv.x;
      kmeta_hw[kk] = kvalid ? mv.y : -1;
    }
    // ---- stage A (weights): lane -> row lane>>1, 32 consecutive k's
    {
      int kb = (lane & 1) * 32;
      unsigned* dst = (unsigned*)&As[lane >> 1][0];
      if (mvld && k4ok && (k0 + 64 <= Kdim)) {
        const float4* wv = (const float4*)(wrow + k0 + kb);
#pragma unroll
        for (int j4 = 0; j4 < 8; ++j4) {
          float4 q = wv[j4];
          dst[(kb >> 1) + j4 * 2 + 0] = pack2h(q.x, q.y);
          dst[(kb >> 1) + j4 * 2 + 1] = pack2h(q.z, q.w);
        }
      } else {
#pragma unroll
        for (int j = 0; j < 32; j += 2) {
          int k = k0 + kb + j;
          int v0 = mvld & ((k < Kdim) ? 1 : 0);
          int v1 = mvld & ((k + 1 < Kdim) ? 1 : 0);
          float f0 = wrow[v0 ? k : 0];
          float f1 = wrow[v1 ? k + 1 : 0];
          f0 = v0 ? f0 : 0.f;
          f1 = v1 ? f1 : 0.f;
          dst[(kb + j) >> 1] = pack2h(f0, f1);
        }
      }
    }
    if (k0 + 64 < Kdim && mvld) __builtin_prefetch(&wrow[k0 + 64], 0, 0);
    __syncthreads();  // kmeta ready for B staging
    // ---- stage B (im2col): lane owns column mycol, k in [khalf*32, +32)
    {
      unsigned* dst = (unsigned*)&Bt[mycol][0];
      const int base = khalf * 32;
#pragma unroll
      for (int j = 0; j < 32; j += 2) {
        float f[2];
#pragma unroll
        for (int u = 0; u < 2; ++u) {
          int kk = base + j + u;
          int hw = kmeta_hw[kk];
          int kh = hw >> 16, kw = hw & 0xffff;
          int ih = col_ih0 + kh;
          int iw = col_iw0 + kw;
          int vld = ((hw >= 0) ? 1 : 0) & col_v &
                    (((unsigned)ih < (unsigned)H) ? 1 : 0) &
                    (((unsigned)iw < (unsigned)W) ? 1 : 0);
          int off = col_off + kmeta_off[kk];
          float fv = x[vld ? off : 0];
          f[u] = vld ? fv : 0.f;
        }
        dst[(base + j) >> 1] = pack2h(f[0], f[1]);
      }
    }
    __syncthreads();
    // ---- two 32-k WMMA steps; both operands read as ds_load_b128 pairs
    const int half_id = lane >> 4;
    const int rm = lane & 15;
#pragma unroll
    for (int s = 0; s < 2; ++s) {
      v16h a, bf;
      const unsigned* asrc = (const unsigned*)&As[rm][0];
      const unsigned* bsrc = (const unsigned*)&Bt[rm][0];
#pragma unroll
      for (int v = 0; v < 8; ++v) {
        int kk = ((v >> 2) * 16) + half_id * 8 + (v & 3) * 2;
        H2U ta, tb;
        ta.u = asrc[(s * 32 + kk) >> 1];
        tb.u = bsrc[(s * 32 + kk) >> 1];
        a[2 * v] = ta.h[0];
        a[2 * v + 1] = ta.h[1];
        bf[2 * v] = tb.h[0];
        bf[2 * v + 1] = tb.h[1];
      }
      acc = __builtin_amdgcn_wmma_f32_16x16x32_f16(false, a, false, bf,
                                                   (short)0, acc, false, false);
    }
  }
  // ---- store C tile: row = v + 8*(lane>=16), col = lane&15
  {
    int col = lane & 15;
    int n = tile_n * 16 + col;
    if (n < Ndim) {
      int b = n / OHW;
      int r2 = n - b * OHW;
      int oh = r2 / OW, ow = r2 - (r2 / OW) * OW;
#pragma unroll
      for (int v = 0; v < 8; ++v) {
        int m = tile_m * 16 + v + 8 * (lane >> 4);
        if (m < Cout) {
          float val = acc[v];
          if (bias) val += bias[m];
          int off = ((b * Cout + m) * OH + oh) * OW + ow;
          if (accumulate) val += y[off];
          if (do_relu) val = val > 0.f ? val : 0.f;
          y[off] = val;
        }
      }
    }
  }
}

// ---------------------------------------------------------------------------
// Fused GroupNorm + ReLU (in place).  One block per (batch, group).
// ---------------------------------------------------------------------------
__global__ void groupnorm_relu_kernel(float* __restrict__ x,
                                      const float* __restrict__ gamma,
                                      const float* __restrict__ beta, int C,
                                      int S, int groups, float eps) {
  int b = blockIdx.x, g = blockIdx.y;
  int cg = C / groups;
  int len = cg * S;
  float* base = x + (b * C + g * cg) * S;
  __shared__ float r1[256];
  __shared__ float r2[256];
  float s = 0.f, s2 = 0.f;
  for (int i = (int)threadIdx.x; i < len; i += (int)blockDim.x) {
    float v = base[i];
    s += v; s2 += v * v;
  }
  r1[threadIdx.x] = s; r2[threadIdx.x] = s2;
  __syncthreads();
  for (int off = (int)blockDim.x / 2; off > 0; off >>= 1) {
    if ((int)threadIdx.x < off) {
      r1[threadIdx.x] += r1[threadIdx.x + off];
      r2[threadIdx.x] += r2[threadIdx.x + off];
    }
    __syncthreads();
  }
  float mean = r1[0] / (float)len;
  float var = r2[0] / (float)len - mean * mean;
  float inv = rsqrtf(var + eps);
  for (int i = (int)threadIdx.x; i < len; i += (int)blockDim.x) {
    int c = i / S;
    float v = (base[i] - mean) * inv * gamma[g * cg + c] + beta[g * cg + c];
    base[i] = v > 0.f ? v : 0.f;
  }
}

// ---------------------------------------------------------------------------
// i==0 index init:  argmax over (Hb*Wb) of mean over channels.
// ---------------------------------------------------------------------------
__global__ void argmax_init_kernel(const float* __restrict__ corr, int N, int C,
                                   int H, int W, int M, int Hb, int Wb,
                                   float* __restrict__ index) {
  unsigned total = (unsigned)(N * H * W * M);
  int sc = H * W * M * Hb * Wb;
  GS_LOOP(idx, total) {
    unsigned t = idx;
    int m = (int)(t % (unsigned)M); t /= (unsigned)M;
    int w = (int)(t % (unsigned)W); t /= (unsigned)W;
    int h = (int)(t % (unsigned)H); t /= (unsigned)H;
    int n = (int)t;
    const float* base = corr + (((n * C * H + h) * W + w) * M + m) * Hb * Wb;
    float best = -3.4e38f;
    int bi = 0;
    for (int p = 0; p < Hb * Wb; ++p) {
      float sum = 0.f;
      for (int c = 0; c < C; ++c) sum += base[c * sc + p];
      if (sum > best) { best = sum; bi = p; }
    }
    index[idx * 2 + 0] = (float)(bi / Wb);
    index[idx * 2 + 1] = (float)(bi % Wb);
  }
}

// ---------------------------------------------------------------------------
// Bilinear cost-volume sampler, fusing local = index + (dyx | di) with zeroed
// center tap.  cost(N,C,H,W,M,9,9), zero padding outside (align_corners=True).
// ---------------------------------------------------------------------------
__global__ void sampler_kernel(const float* __restrict__ corr,
                               const float* __restrict__ index,
                               const float* __restrict__ di,
                               float* __restrict__ cost, int N, int C, int H,
                               int W, int M, int Hb, int Wb) {
  const int T = 81;
  unsigned total = (unsigned)(N * C * H * W * M) * (unsigned)T;
  GS_LOOP(idx, total) {
    unsigned t = idx;
    int p = (int)(t % (unsigned)T); t /= (unsigned)T;
    int m = (int)(t % (unsigned)M); t /= (unsigned)M;
    int w = (int)(t % (unsigned)W); t /= (unsigned)W;
    int h = (int)(t % (unsigned)H); t /= (unsigned)H;
    int c = (int)(t % (unsigned)C); t /= (unsigned)C;
    int n = (int)t;
    int q = ((n * H + h) * W + w) * M + m;
    float gy = index[q * 2 + 0], gx = index[q * 2 + 1];
    if (di) {
      if (p != 40) {  // center (r,r) forced to zero offset
        gy += di[(q * T + p) * 2 + 0];
        gx += di[(q * T + p) * 2 + 1];
      }
    } else {
      gy += (float)(p / 9 - 4);
      gx += (float)(p % 9 - 4);
    }
    float x0f = floorf(gx), y0f = floorf(gy);
    int x0 = (int)x0f, y0 = (int)y0f;
    float wx = gx - x0f, wy = gy - y0f;
    const float* img =
        corr + (((n * C + c) * H + h) * W + w) * M * Hb * Wb + m * Hb * Wb;
    float v00 = (x0 >= 0 && x0 < Wb && y0 >= 0 && y0 < Hb) ? img[y0 * Wb + x0] : 0.f;
    float v10 = (x0 + 1 >= 0 && x0 + 1 < Wb && y0 >= 0 && y0 < Hb) ? img[y0 * Wb + x0 + 1] : 0.f;
    float v01 = (x0 >= 0 && x0 < Wb && y0 + 1 >= 0 && y0 + 1 < Hb) ? img[(y0 + 1) * Wb + x0] : 0.f;
    float v11 = (x0 + 1 >= 0 && x0 + 1 < Wb && y0 + 1 >= 0 && y0 + 1 < Hb) ? img[(y0 + 1) * Wb + x0 + 1] : 0.f;
    cost[idx] = v00 * (1.f - wx) * (1.f - wy) + v10 * wx * (1.f - wy) +
                v01 * (1.f - wx) * wy + v11 * wx * wy;
  }
}

__global__ void mean_last_kernel(const float* __restrict__ in,
                                 float* __restrict__ out, unsigned n_out, int M) {
  float inv = 1.f / (float)M;
  GS_LOOP(idx, n_out) {
    float s = 0.f;
    for (int m = 0; m < M; ++m) s += in[idx * M + m];
    out[idx] = s * inv;
  }
}

__global__ void add2_kernel(const float* __restrict__ a, const float* __restrict__ b,
                            float* __restrict__ c, unsigned n) {
  GS_LOOP(i, n) c[i] = a[i] + b[i];
}

// resize over (ha,wa) for layout (B,C,Hi,Wi,Nn), plus residual add.
__global__ void resize5_add_kernel(const float* __restrict__ in,
                                   const float* __restrict__ add,
                                   float* __restrict__ out, int B, int C, int Hi,
                                   int Wi, int Nn, int Ho, int Wo) {
  unsigned total = (unsigned)(B * C * Ho * Wo * Nn);
  GS_LOOP(idx, total) {
    unsigned t = idx;
    int nn = (int)(t % (unsigned)Nn); t /= (unsigned)Nn;
    int wo = (int)(t % (unsigned)Wo); t /= (unsigned)Wo;
    int ho = (int)(t % (unsigned)Ho); t /= (unsigned)Ho;
    int c = (int)(t % (unsigned)C); t /= (unsigned)C;
    int b = (int)t;
    float ys = (Ho > 1) ? (float)ho * (float)(Hi - 1) / (float)(Ho - 1) : 0.f;
    float xs = (Wo > 1) ? (float)wo * (float)(Wi - 1) / (float)(Wo - 1) : 0.f;
    int y0 = (int)floorf(ys), x0 = (int)floorf(xs);
    int y1 = y0 + 1 < Hi ? y0 + 1 : Hi - 1;
    int x1 = x0 + 1 < Wi ? x0 + 1 : Wi - 1;
    float wy = ys - (float)y0, wx = xs - (float)x0;
    const float* base = in + (b * C + c) * Hi * Wi * Nn;
    float v00 = base[(y0 * Wi + x0) * Nn + nn];
    float v01 = base[(y0 * Wi + x1) * Nn + nn];
    float v10 = base[(y1 * Wi + x0) * Nn + nn];
    float v11 = base[(y1 * Wi + x1) * Nn + nn];
    float v = v00 * (1 - wy) * (1 - wx) + v01 * (1 - wy) * wx +
              v10 * wy * (1 - wx) + v11 * wy * wx;
    out[idx] = v + add[idx];
  }
}

// plain NCHW bilinear resize (align-corners linspace, as resize_ac)
__global__ void resize_nchw_kernel(const float* __restrict__ in,
                                   float* __restrict__ out, int B, int C, int Hi,
                                   int Wi, int Ho, int Wo) {
  unsigned total = (unsigned)(B * C * Ho * Wo);
  GS_LOOP(idx, total) {
    unsigned t = idx;
    int wo = (int)(t % (unsigned)Wo); t /= (unsigned)Wo;
    int ho = (int)(t % (unsigned)Ho); t /= (unsigned)Ho;
    int c = (int)(t % (unsigned)C); t /= (unsigned)C;
    int b = (int)t;
    float ys = (Ho > 1) ? (float)ho * (float)(Hi - 1) / (float)(Ho - 1) : 0.f;
    float xs = (Wo > 1) ? (float)wo * (float)(Wi - 1) / (float)(Wo - 1) : 0.f;
    int y0 = (int)floorf(ys), x0 = (int)floorf(xs);
    int y1 = y0 + 1 < Hi ? y0 + 1 : Hi - 1;
    int x1 = x0 + 1 < Wi ? x0 + 1 : Wi - 1;
    float wy = ys - (float)y0, wx = xs - (float)x0;
    const float* base = in + (b * C + c) * Hi * Wi;
    float v = base[y0 * Wi + x0] * (1 - wy) * (1 - wx) +
              base[y0 * Wi + x1] * (1 - wy) * wx +
              base[y1 * Wi + x0] * wy * (1 - wx) +
              base[y1 * Wi + x1] * wy * wx;
    out[idx] = v;
  }
}

// ===========================================================================
// Host-side orchestration
// ===========================================================================
namespace {

struct Arena {
  char* base; size_t off, cap;
  float* alloc(long long nelem) {
    size_t bytes = ((size_t)nelem * sizeof(float) + 255) & ~(size_t)255;
    float* p = (float*)(base + off);
    off += bytes;
    return p;
  }
  size_t mark() const { return off; }
  void reset(size_t m) { off = m; }
};

inline unsigned gs_blocks(long long total) {
  long long b = (total + 255) / 256;
  if (b > 1048576) b = 1048576;
  if (b < 1) b = 1;
  return (unsigned)b;
}

void gather7(hipStream_t st, const float* in, float* out, int d0, int d1, int d2,
             int d3, int d4, int d5, int d6, long long s0, long long s1,
             long long s2, long long s3, long long s4, long long s5,
             long long s6, int acc) {
  long long total = (long long)d0 * d1 * d2 * d3 * d4 * d5 * d6;
  gather7_kernel<<<gs_blocks(total), 256, 0, st>>>(
      in, out, (unsigned)total, d1, d2, d3, d4, d5, d6, (int)s0, (int)s1,
      (int)s2, (int)s3, (int)s4, (int)s5, (int)s6, acc);
}

void conv2d_h(hipStream_t st, Arena& A, const float* x, int B, int Cin, int H,
              int W, const float* w, const float* bias, int Cout, int K, int s,
              int pad, float* y, int acc, int relu) {
  int OH = (H + 2 * pad - K) / s + 1;
  int OW = (W + 2 * pad - K) / s + 1;
  int Kdim = Cin * K * K;
  long long Nd = (long long)B * OH * OW;
  // per-launch kmeta table (stream order makes arena recycling safe)
  int2* km = (int2*)A.alloc((long long)Kdim * 2);
  kmeta_build_kernel<<<gs_blocks(Kdim), 256, 0, st>>>(km, Kdim, K * K, K,
                                                      H * W, W);
  dim3 grid((unsigned)((Nd + 15) / 16), (unsigned)((Cout + 15) / 16));
  conv2d_wmma_kernel<<<grid, 32, 0, st>>>(x, w, bias, y, km, B, Cin, H, W,
                                          Cout, K, s, pad, OH, OW, acc, relu);
}

void groupnorm_h(hipStream_t st, float* x, int B, int C, long long S,
                 const float* g, const float* bt) {
  dim3 grid((unsigned)B, 4);
  groupnorm_relu_kernel<<<grid, 256, 0, st>>>(x, g, bt, C, (int)S, 4, 1e-5f);
}

// FC einsum via WMMA 1x1 conv:  out(N,H,W,M,O) = sum_c agg(N,c,H,W,M) wt(O,c)
// agg viewed as image (N, 128, H, W*M); conv 1x1; then transpose O innermost.
void fc_h(hipStream_t st, Arena& A, const float* agg, const float* wt, int N,
          int Cc, int H, int W, int M, int O, float* out) {
  long long HWM = (long long)H * W * M;
  float* tmp = A.alloc((long long)N * O * HWM);  // (N,O,H,W*M)
  conv2d_h(st, A, agg, N, Cc, H, W * M, wt, nullptr, O, 1, 1, 0, tmp, 0, 0);
  gather7(st, tmp, out, N, H, W, M, O, 1, 1, (long long)O * HWM,
          (long long)W * M, (long long)M, 1LL, HWM, 0LL, 0LL, 0);
}

// conv4d (center pivot): out(b,oc,ha,wa,n,ohb,owb), x(b,c,ha,wa,n,hb,wb)
void conv4d_h(hipStream_t st, Arena& A, const float* x, int b, int c, int ha,
              int wa, int n, int hb, int wb, const float* w1, const float* b1_,
              const float* w2, const float* b2_, int oc, int k, int s,
              float* out, int* ohb_p, int* owb_p) {
  int pad = k / 2;
  int ohb = (hb + 2 * pad - k) / s + 1;
  int owb = (wb + 2 * pad - k) / s + 1;
  *ohb_p = ohb; *owb_p = owb;
  size_t mk = A.mark();
  // x strides
  long long swb = 1, shb = wb, sn = (long long)hb * wb, swa = sn * n,
            sha = swa * wa, sc = sha * ha, sb = sc * c;
  // ---- branch 2: (b,ha,wa,n) batch, (hb,wb) spatial
  long long B2 = (long long)b * ha * wa * n;
  float* x2 = A.alloc(B2 * c * hb * wb);
  gather7(st, x, x2, b, ha, wa, n, c, hb, wb, sb, sha, swa, sn, sc, shb, swb, 0);
  float* o2 = A.alloc(B2 * oc * ohb * owb);
  conv2d_h(st, A, x2, (int)B2, c, hb, wb, w2, b2_, oc, k, s, pad, o2, 0, 0);
  // scatter o2 (b,ha,wa,n,oc,ohb,owb) -> out (b,oc,ha,wa,n,ohb,owb)
  long long t_owb = 1, t_ohb = owb, t_oc = (long long)ohb * owb, t_n = t_oc * oc,
            t_wa = t_n * n, t_ha = t_wa * wa, t_b = t_ha * ha;
  gather7(st, o2, out, b, oc, ha, wa, n, ohb, owb, t_b, t_oc, t_ha, t_wa, t_n,
          t_ohb, t_owb, 0);
  // ---- branch 1: prune (hb,wb) by ::s, conv over (ha,wa)
  int phb = (hb + s - 1) / s, pwb = (wb + s - 1) / s;  // == ohb,owb for this net
  long long B1 = (long long)b * n * phb * pwb;
  float* x1 = A.alloc(B1 * c * ha * wa);
  gather7(st, x, x1, b, n, phb, pwb, c, ha, wa, sb, sn, shb * s, swb * s, sc,
          sha, swa, 0);
  float* o1 = A.alloc(B1 * oc * ha * wa);
  conv2d_h(st, A, x1, (int)B1, c, ha, wa, w1, b1_, oc, k, 1, pad, o1, 0, 0);
  // add o1 (b,n,phb,pwb,oc,ha,wa) into out (b,oc,ha,wa,n,ohb,owb)
  long long u_wa = 1, u_ha = wa, u_oc = (long long)ha * wa, u_pwb = u_oc * oc,
            u_phb = u_pwb * pwb, u_n = u_phb * phb, u_b = u_n * n;
  gather7(st, o1, out, b, oc, ha, wa, n, ohb, owb, u_b, u_oc, u_ha, u_wa, u_n,
          u_phb, u_pwb, 1);
  A.reset(mk);
}

void run_block_h(hipStream_t st, Arena& A, const float* x, int b, int c, int ha,
                 int wa, int n, int hb, int wb, const float* const* prm,
                 const int* ks, const int* ss, const int* ocs, int nl,
                 float* last_out) {
  const float* cur = x;
  int cc = c, chb = hb, cwb = wb;
  size_t mk = A.mark();
  for (int L = 0; L < nl; ++L) {
    int oc = ocs[L], k = ks[L], s = ss[L];
    int pad = k / 2;
    int ohb = (chb + 2 * pad - k) / s + 1;
    int owb = (cwb + 2 * pad - k) / s + 1;
    long long osz = (long long)b * oc * ha * wa * n * ohb * owb;
    float* out = (L == nl - 1) ? last_out : A.alloc(osz);
    int oh2, ow2;
    conv4d_h(st, A, cur, b, cc, ha, wa, n, chb, cwb, prm[L * 6 + 0],
             prm[L * 6 + 1], prm[L * 6 + 2], prm[L * 6 + 3], oc, k, s, out,
             &oh2, &ow2);
    groupnorm_h(st, out, b, oc, (long long)ha * wa * n * oh2 * ow2,
                prm[L * 6 + 4], prm[L * 6 + 5]);
    cur = out; cc = oc; chb = oh2; cwb = ow2;
  }
  A.reset(mk);
}

void run_layer_h(hipStream_t st, Arena& A, const float* corr, int N, int C,
                 int H, int W, int M, int Hb, int Wb, const float* const* prm,
                 const int* ks, const int* ss, const float* dw, const float* sw,
                 const float* sb_, const float* fw, float* sqz_out,
                 float* sals_out, float* inds_out) {
  static const int ocs[3] = {16, 64, 128};
  long long Q = (long long)N * H * W * M;
  float* idx = A.alloc(Q * 2);
  float* idx2 = A.alloc(Q * 2);
  float* delta = A.alloc(Q * 2);
  float* di = A.alloc(Q * 162);
  float* sq_mean = A.alloc((long long)N * 128 * H * W);
  size_t mk = A.mark();
  for (int i = 0; i < 5; ++i) {
    A.reset(mk);
    const float* duse = nullptr;
    if (i == 0) {
      argmax_init_kernel<<<gs_blocks(Q), 256, 0, st>>>(corr, N, C, H, W, M, Hb,
                                                       Wb, idx);
    } else {
      fc_h(st, A, sqz_out, fw, N, 128, H, W, M, 162, di);
      add2_kernel<<<gs_blocks(Q * 2), 256, 0, st>>>(idx, delta, idx2,
                                                    (unsigned)(Q * 2));
      float* t = idx; idx = idx2; idx2 = t;
      duse = di;
    }
    hipMemcpyAsync(inds_out + (long long)i * Q * 2, idx,
                   (size_t)(Q * 2) * sizeof(float), hipMemcpyDeviceToDevice, st);
    long long csz = (long long)N * C * H * W * M * 81;
    float* cost = A.alloc(csz);
    sampler_kernel<<<gs_blocks(csz), 256, 0, st>>>(corr, idx, duse, cost, N, C,
                                                   H, W, M, Hb, Wb);
    run_block_h(st, A, cost, N, C, H, W, M, 9, 9, prm, ks, ss, ocs, 3, sqz_out);
    mean_last_kernel<<<gs_blocks((long long)N * 128 * H * W), 256, 0, st>>>(
        sqz_out, sq_mean, (unsigned)(N * 128 * H * W), M);
    conv2d_h(st, A, sq_mean, N, 128, H, W, sw, sb_, 1, 3, 1, 1,
             sals_out + (long long)i * N * H * W, 0, 0);
    fc_h(st, A, sqz_out, dw, N, 128, H, W, M, 2, delta);
  }
}

}  // namespace

// ---------------------------------------------------------------------------
// kernel_launch: inputs in setup_inputs() insertion order (109 flat tensors):
//  0 corr4  1 corr3  2 corr2
//  3..20 enc4 (3 layers x [w1,b1,w2,b2,g,bt])   21..38 enc3   39..56 enc2
//  57..74 enc43   75..92 enc32
//  93 d3w 94 d2w 95 s3w 96 s3b 97 s2w 98 s2b 99 f3w 100 f2w
//  101..104 dec1p [w1,b1,w2,b2]   105..108 dec2p
// d_out: [d(4,128,64,64) | sals3(5,4,1,16,16) | sals2(5,4,1,32,32) |
//         inds3(5,4,16,16,4,2) | inds2(5,4,32,32,4,2)]
// ---------------------------------------------------------------------------
extern "C" void kernel_launch(void* const* d_in, const int* in_sizes, int n_in,
                              void* d_out, int out_size, void* d_ws,
                              size_t ws_size, hipStream_t stream) {
  (void)in_sizes; (void)n_in; (void)out_size;
  auto P = [&](int i) { return (const float*)d_in[i]; };
  Arena A{(char*)d_ws, 0, ws_size};

  const float* enc4[18];  for (int i = 0; i < 18; ++i) enc4[i] = P(3 + i);
  const float* enc3[18];  for (int i = 0; i < 18; ++i) enc3[i] = P(21 + i);
  const float* enc2[18];  for (int i = 0; i < 18; ++i) enc2[i] = P(39 + i);
  const float* enc43[18]; for (int i = 0; i < 18; ++i) enc43[i] = P(57 + i);
  const float* enc32[18]; for (int i = 0; i < 18; ++i) enc32[i] = P(75 + i);

  float* out = (float*)d_out;
  float* d_img = out;                       // 4*128*64*64 = 2097152
  float* sals3_o = out + 2097152;           // 5120
  float* sals2_o = sals3_o + 5120;          // 20480
  float* inds3_o = sals2_o + 20480;         // 40960
  float* inds2_o = inds3_o + 40960;         // 163840

  static const int ocsE[3] = {16, 64, 128};
  static const int ocsM[3] = {128, 128, 128};

  // persistent buffers
  float* sqz4 = A.alloc(4LL * 128 * 8 * 8 * 4);
  float* sqz3 = A.alloc(4LL * 128 * 16 * 16 * 4);
  float* sqz2 = A.alloc(4LL * 128 * 32 * 32 * 4);
  float* mix43 = A.alloc(4LL * 128 * 16 * 16 * 4);
  float* mix432 = A.alloc(4LL * 128 * 32 * 32 * 4);
  float* tmp43 = A.alloc(4LL * 128 * 16 * 16 * 4);
  float* tmp32 = A.alloc(4LL * 128 * 32 * 32 * 4);
  float* e_buf = A.alloc(4LL * 128 * 32 * 32);
  float* t1 = A.alloc(4LL * 128 * 32 * 32);
  float* t2 = A.alloc(4LL * 128 * 32 * 32);
  float* t3 = A.alloc(4LL * 128 * 64 * 64);
  float* t4 = A.alloc(4LL * 128 * 64 * 64);
  size_t mk = A.mark();

  // ---- sqz4 = run_block(corr4, enc4, [3,3,3], [2,2,2])
  {
    int ks[3] = {3, 3, 3}, ss[3] = {2, 2, 2};
    run_block_h(stream, A, P(0), 4, 3, 8, 8, 4, 8, 8, enc4, ks, ss, ocsE, 3, sqz4);
  }
  A.reset(mk);
  // ---- layer 3
  {
    int ks[3] = {5, 3, 3}, ss[3] = {4, 2, 2};
    run_layer_h(stream, A, P(1), 4, 4, 16, 16, 4, 16, 16, enc3, ks, ss, P(93),
                P(95), P(96), P(99), sqz3, sals3_o, inds3_o);
  }
  A.reset(mk);
  // ---- layer 2
  {
    int ks[3] = {5, 5, 3}, ss[3] = {4, 4, 2};
    run_layer_h(stream, A, P(2), 4, 4, 32, 32, 4, 32, 32, enc2, ks, ss, P(94),
                P(97), P(98), P(100), sqz2, sals2_o, inds2_o);
  }
  A.reset(mk);
  // ---- mix43 = run_block(interp(sqz4 -> 16x16) + sqz3, enc43)
  resize5_add_kernel<<<gs_blocks(4LL * 128 * 16 * 16 * 4), 256, 0, stream>>>(
      sqz4, sqz3, tmp43, 4, 128, 8, 8, 4, 16, 16);
  {
    int ks[3] = {3, 3, 3}, ss[3] = {1, 1, 1};
    run_block_h(stream, A, tmp43, 4, 128, 16, 16, 4, 1, 1, enc43, ks, ss, ocsM,
                3, mix43);
  }
  A.reset(mk);
  // ---- mix432 = run_block(interp(mix43 -> 32x32) + sqz2, enc32)
  resize5_add_kernel<<<gs_blocks(4LL * 128 * 32 * 32 * 4), 256, 0, stream>>>(
      mix43, sqz2, tmp32, 4, 128, 16, 16, 4, 32, 32);
  {
    int ks[3] = {3, 3, 3}, ss[3] = {1, 1, 1};
    run_block_h(stream, A, tmp32, 4, 128, 32, 32, 4, 1, 1, enc32, ks, ss, ocsM,
                3, mix432);
  }
  A.reset(mk);
  // ---- decoder
  mean_last_kernel<<<gs_blocks(4LL * 128 * 32 * 32), 256, 0, stream>>>(
      mix432, e_buf, (unsigned)(4 * 128 * 32 * 32), 4);
  conv2d_h(stream, A, e_buf, 4, 128, 32, 32, P(101), P(102), 128, 3, 1, 1, t1, 0, 1);
  conv2d_h(stream, A, t1, 4, 128, 32, 32, P(103), P(104), 128, 3, 1, 1, t2, 0, 1);
  resize_nchw_kernel<<<gs_blocks(4LL * 128 * 64 * 64), 256, 0, stream>>>(
      t2, t3, 4, 128, 32, 32, 64, 64);
  conv2d_h(stream, A, t3, 4, 128, 64, 64, P(105), P(106), 128, 3, 1, 1, t4, 0, 1);
  conv2d_h(stream, A, t4, 4, 128, 64, 64, P(107), P(108), 128, 3, 1, 1, d_img, 0, 0);
}
